// HybridSchNetMPNN_73254962200853
// MI455X (gfx1250) — matrix operand order repaired and checked by reading, not compiled, not run on hardware
//
#include <hip/hip_runtime.h>
#include <hip/hip_bf16.h>

typedef __attribute__((ext_vector_type(16))) _Float16 v16h;
typedef __attribute__((ext_vector_type(8)))  _Float16 v8h;
typedef __attribute__((ext_vector_type(8)))  float    v8f;

#define NN 20000
#define EE 640000
#define GG 64
#define HID 128

__device__ inline v8f wmma16(v16h a, v16h b, v8f c) {
    return __builtin_amdgcn_wmma_f32_16x16x32_f16(
        false, a, false, b, (short)0, c, false, false);
}

__device__ inline v8f splat8(float v) { v8f c = {v,v,v,v,v,v,v,v}; return c; }

__device__ inline void fatomic_add(float* p, float v) {
    __hip_atomic_fetch_add(p, v, __ATOMIC_RELAXED, __HIP_MEMORY_SCOPE_AGENT);
}

// ---- A-fragment helpers (ISA 7.12.2 16-bit A 16x32 layout) ----
// lanes 0-15: row=lane,   K chunks {kt*32+0..7, kt*32+16..23}
// lanes16-31: row=lane-16,K chunks {kt*32+8..15, kt*32+24..31}

__device__ inline v16h ldsA(const _Float16* tile, int lane, int kt) {
    int row  = lane & 15;
    int koff = kt * 32 + ((lane & 16) ? 8 : 0);
    const _Float16* p = tile + row * 136 + koff;
    v8h lo = *(const v8h*)(p);
    v8h hi = *(const v8h*)(p + 16);
    v16h a;
#pragma unroll
    for (int i = 0; i < 8; ++i) { a[i] = lo[i]; a[i + 8] = hi[i]; }
    return a;
}

__device__ inline void storeD(_Float16* tile, int lane, int nt, v8f d) {
    int col   = nt * 16 + (lane & 15);
    int rbase = (lane & 16) ? 8 : 0;
#pragma unroll
    for (int r = 0; r < 8; ++r)
        tile[(rbase + r) * 136 + col] = (_Float16)d[r];
}

// ---- Weight packing into B-fragment order (32x16 f16 B tiles) ----
__global__ void pack_b(const float* __restrict__ W, int K, int KT,
                       _Float16* __restrict__ dst) {
    int i = blockIdx.x * blockDim.x + threadIdx.x;
    int total = KT * 8 * 32 * 16;
    if (i >= total) return;
    int e    = i & 15;
    int lane = (i >> 4) & 31;
    int nt   = (i >> 9) & 7;
    int kt   = i >> 12;
    int k = kt * 32 + ((lane >= 16) ? 8 : 0) + e + ((e >= 8) ? 8 : 0);
    int n = nt * 16 + (lane & 15);
    dst[i] = (k < K) ? (_Float16)W[(size_t)k * 128 + n] : (_Float16)0.f;
}

// ---- Node-side matmul: out = relu(A[N,K] @ W[K,128] + b), WMMA ----
template <int K, int NW>
__global__ __launch_bounds__(NW * 32) void node_update_kernel(
    const float* __restrict__ A, const _Float16* __restrict__ Bw,
    const float* __restrict__ bias, float* __restrict__ out, int ntiles) {
    int wid = threadIdx.x >> 5, lane = threadIdx.x & 31;
    int tile = blockIdx.x * NW + wid;
    if (tile >= ntiles) return;  // uniform per wave, EXEC stays full
    constexpr int KT = K / 32;
    const float* row = A + ((size_t)tile * 16 + (lane & 15)) * K;
    int kadj = (lane & 16) ? 8 : 0;
    v16h a[KT];
#pragma unroll
    for (int kt = 0; kt < KT; ++kt) {
        int ko = kt * 32 + kadj;
#pragma unroll
        for (int i = 0; i < 8; ++i) {
            a[kt][i]     = (_Float16)row[ko + i];
            a[kt][i + 8] = (_Float16)row[ko + 16 + i];
        }
    }
    const v16h* B = (const v16h*)Bw;
    size_t obase = (size_t)tile * 16 * 128;
    int f = (lane & 15);
    int rbase = (lane & 16) ? 8 : 0;
#pragma unroll
    for (int nt = 0; nt < 8; ++nt) {
        v8f c = splat8(bias[nt * 16 + f]);
#pragma unroll
        for (int kt = 0; kt < KT; ++kt)
            c = wmma16(a[kt], B[(kt * 8 + nt) * 32 + lane], c);
#pragma unroll
        for (int r = 0; r < 8; ++r)
            out[obase + (size_t)(rbase + r) * 128 + nt * 16 + f] =
                fmaxf(c[r], 0.f);
    }
}

// ---- Fused SchNet edge layer: 2 M-tiles (32 edges) per wave ----
// e=relu(attr@Wee+bee); t1=silu(e@f1+b1); w=t1@f2+b2;
// msg = w*(h[src]+h[dst]); atomic scatter-add into agg[dst].
template <int NW>
__global__ __launch_bounds__(NW * 32) void schnet_edge_kernel(
    const float* __restrict__ edge_attr, const int* __restrict__ eix,
    const float* __restrict__ h, float* __restrict__ agg,
    const _Float16* __restrict__ Bee, const float* __restrict__ bee,
    const _Float16* __restrict__ Bf1, const float* __restrict__ bf1,
    const _Float16* __restrict__ Bf2, const float* __restrict__ bf2) {
    __shared__ __align__(16) _Float16 lds[NW * 2 * 2176];
    int wid = threadIdx.x >> 5, lane = threadIdx.x & 31;
    int f = lane & 15;
    long base = ((long)blockIdx.x * NW + wid) * 32;     // 32 edges / wave
    _Float16* t0 = lds + wid * 2 * 2176;
    _Float16* t1 = t0 + 2176;

    // ---- edge indices first + prefetch h rows (hidden under matmuls) ----
    int sidx[16], didx[16];
    {
        long ebase = base + ((lane & 16) ? 8 : 0);
#pragma unroll
        for (int r = 0; r < 8; ++r) {
            long e0 = ebase + r;
            sidx[r]     = eix[2 * e0];
            didx[r]     = eix[2 * ((long)EE + e0)];
            sidx[r + 8] = eix[2 * (e0 + 16)];
            didx[r + 8] = eix[2 * ((long)EE + e0 + 16)];
        }
#pragma unroll
        for (int r = 0; r < 16; ++r) {   // lanes 0..15 stride 32B -> cover 512B row
            __builtin_prefetch(h + (size_t)sidx[r] * 128 + f * 8, 0, 3);
            __builtin_prefetch(h + (size_t)didx[r] * 128 + f * 8, 0, 3);
        }
    }

    // ---- Stage 0: edge embedding for both tiles (K=16 padded to 32) ----
    {
        const float* p0 = edge_attr + (size_t)(base + f) * 16 + ((lane & 16) ? 8 : 0);
        const float* p1 = p0 + 16 * 16;
        v16h a0, a1;
#pragma unroll
        for (int i = 0; i < 8; ++i) {
            a0[i] = (_Float16)p0[i]; a0[i + 8] = (_Float16)0.f;
            a1[i] = (_Float16)p1[i]; a1[i + 8] = (_Float16)0.f;
        }
        const v16h* B = (const v16h*)Bee;
#pragma unroll
        for (int nt = 0; nt < 8; ++nt) {
            v16h b = B[nt * 32 + lane];
            v8f c0 = splat8(bee[nt * 16 + f]);
            v8f c1 = c0;
            c0 = wmma16(a0, b, c0);
            c1 = wmma16(a1, b, c1);
#pragma unroll
            for (int r = 0; r < 8; ++r) {
                c0[r] = fmaxf(c0[r], 0.f);
                c1[r] = fmaxf(c1[r], 0.f);
            }
            storeD(t0, lane, nt, c0);
            storeD(t1, lane, nt, c1);
        }
    }
    asm volatile("s_wait_dscnt 0" ::: "memory");
    v16h ea0[4], ea1[4];
#pragma unroll
    for (int kt = 0; kt < 4; ++kt) {
        ea0[kt] = ldsA(t0, lane, kt);
        ea1[kt] = ldsA(t1, lane, kt);
    }

    // ---- Stage 1: t1 = silu(e @ f1 + b1); reuse the same LDS tiles ----
    asm volatile("s_wait_dscnt 0" ::: "memory");
    {
        const v16h* B = (const v16h*)Bf1;
#pragma unroll
        for (int nt = 0; nt < 8; ++nt) {
            v8f c0 = splat8(bf1[nt * 16 + f]);
            v8f c1 = c0;
#pragma unroll
            for (int kt = 0; kt < 4; ++kt) {
                v16h b = B[(kt * 8 + nt) * 32 + lane];
                c0 = wmma16(ea0[kt], b, c0);
                c1 = wmma16(ea1[kt], b, c1);
            }
#pragma unroll
            for (int r = 0; r < 8; ++r) {
                float x0 = c0[r]; c0[r] = x0 / (1.0f + __expf(-x0));
                float x1 = c1[r]; c1[r] = x1 / (1.0f + __expf(-x1));
            }
            storeD(t0, lane, nt, c0);
            storeD(t1, lane, nt, c1);
        }
    }
    asm volatile("s_wait_dscnt 0" ::: "memory");
    v16h ta0[4], ta1[4];
#pragma unroll
    for (int kt = 0; kt < 4; ++kt) {
        ta0[kt] = ldsA(t0, lane, kt);
        ta1[kt] = ldsA(t1, lane, kt);
    }

    // ---- Stage 2: w = t1 @ f2 + b2, fused gather/scale/scatter ----
    {
        const v16h* B = (const v16h*)Bf2;
#pragma unroll
        for (int nt = 0; nt < 8; ++nt) {
            v8f c0 = splat8(bf2[nt * 16 + f]);
            v8f c1 = c0;
#pragma unroll
            for (int kt = 0; kt < 4; ++kt) {
                v16h b = B[(kt * 8 + nt) * 32 + lane];
                c0 = wmma16(ta0[kt], b, c0);
                c1 = wmma16(ta1[kt], b, c1);
            }
            int col = nt * 16 + f;
#pragma unroll
            for (int r = 0; r < 8; ++r) {
                float hs0 = h[(size_t)sidx[r] * 128 + col] +
                            h[(size_t)didx[r] * 128 + col];
                fatomic_add(&agg[(size_t)didx[r] * 128 + col], c0[r] * hs0);
                float hs1 = h[(size_t)sidx[r + 8] * 128 + col] +
                            h[(size_t)didx[r + 8] * 128 + col];
                fatomic_add(&agg[(size_t)didx[r + 8] * 128 + col], c1[r] * hs1);
            }
        }
    }
}

// ---- Fused MPNN edge layer: 2 M-tiles per wave ----
// ea = e@We+b; scatter [h[src] | ea] into agg2[dst].
template <int NW>
__global__ __launch_bounds__(NW * 32) void mpnn_edge_kernel(
    const float* __restrict__ edge_attr, const int* __restrict__ eix,
    const float* __restrict__ h, float* __restrict__ agg2,
    const _Float16* __restrict__ Bee, const float* __restrict__ bee,
    const _Float16* __restrict__ Bme, const float* __restrict__ bme) {
    __shared__ __align__(16) _Float16 lds[NW * 2 * 2176];
    int wid = threadIdx.x >> 5, lane = threadIdx.x & 31;
    int f = lane & 15;
    long base = ((long)blockIdx.x * NW + wid) * 32;
    _Float16* t0 = lds + wid * 2 * 2176;
    _Float16* t1 = t0 + 2176;

    int sidx[16], didx[16];
    {
        long ebase = base + ((lane & 16) ? 8 : 0);
#pragma unroll
        for (int r = 0; r < 8; ++r) {
            long e0 = ebase + r;
            sidx[r]     = eix[2 * e0];
            didx[r]     = eix[2 * ((long)EE + e0)];
            sidx[r + 8] = eix[2 * (e0 + 16)];
            didx[r + 8] = eix[2 * ((long)EE + e0 + 16)];
        }
#pragma unroll
        for (int r = 0; r < 16; ++r) {
            __builtin_prefetch(h + (size_t)sidx[r] * 128 + f * 8, 0, 3);
            __builtin_prefetch(h + (size_t)didx[r] * 128 + f * 8, 0, 3);
        }
    }

    {   // edge embedding for both tiles
        const float* p0 = edge_attr + (size_t)(base + f) * 16 + ((lane & 16) ? 8 : 0);
        const float* p1 = p0 + 16 * 16;
        v16h a0, a1;
#pragma unroll
        for (int i = 0; i < 8; ++i) {
            a0[i] = (_Float16)p0[i]; a0[i + 8] = (_Float16)0.f;
            a1[i] = (_Float16)p1[i]; a1[i + 8] = (_Float16)0.f;
        }
        const v16h* B = (const v16h*)Bee;
#pragma unroll
        for (int nt = 0; nt < 8; ++nt) {
            v16h b = B[nt * 32 + lane];
            v8f c0 = splat8(bee[nt * 16 + f]);
            v8f c1 = c0;
            c0 = wmma16(a0, b, c0);
            c1 = wmma16(a1, b, c1);
#pragma unroll
            for (int r = 0; r < 8; ++r) {
                c0[r] = fmaxf(c0[r], 0.f);
                c1[r] = fmaxf(c1[r], 0.f);
            }
            storeD(t0, lane, nt, c0);
            storeD(t1, lane, nt, c1);
        }
    }
    asm volatile("s_wait_dscnt 0" ::: "memory");
    v16h ea0[4], ea1[4];
#pragma unroll
    for (int kt = 0; kt < 4; ++kt) {
        ea0[kt] = ldsA(t0, lane, kt);
        ea1[kt] = ldsA(t1, lane, kt);
    }

    const v16h* B = (const v16h*)Bme;
#pragma unroll
    for (int nt = 0; nt < 8; ++nt) {
        v8f c0 = splat8(bme[nt * 16 + f]);
        v8f c1 = c0;
#pragma unroll
        for (int kt = 0; kt < 4; ++kt) {
            v16h b = B[(kt * 8 + nt) * 32 + lane];
            c0 = wmma16(ea0[kt], b, c0);
            c1 = wmma16(ea1[kt], b, c1);
        }
        int col = nt * 16 + f;
#pragma unroll
        for (int r = 0; r < 8; ++r) {
            size_t d0 = (size_t)didx[r] * 256;
            fatomic_add(&agg2[d0 + 128 + col], c0[r]);
            fatomic_add(&agg2[d0 + col], h[(size_t)sidx[r] * 128 + col]);
            size_t d1 = (size_t)didx[r + 8] * 256;
            fatomic_add(&agg2[d1 + 128 + col], c1[r]);
            fatomic_add(&agg2[d1 + col], h[(size_t)sidx[r + 8] * 128 + col]);
        }
    }
}

// ---- Global mean-pool accumulation ----
__global__ void pool_kernel(const float* __restrict__ h,
                            const int* __restrict__ batch,
                            float* __restrict__ pooled, float* __restrict__ cnt) {
    int i = blockIdx.x * blockDim.x + threadIdx.x;
    if (i >= NN * 128) return;
    int n = i >> 7, f = i & 127;
    int g = batch[2 * n];  // int64 low dword
    fatomic_add(&pooled[g * 128 + f], h[i]);
    if (f == 0) fatomic_add(&cnt[g], 1.0f);
}

// ---- Head: out = relu(mean @ fc1 + b1) @ fc2 + b2 ----
__global__ void head_kernel(const float* __restrict__ pooled,
                            const float* __restrict__ cnt,
                            const float* __restrict__ fc1w, const float* __restrict__ fc1b,
                            const float* __restrict__ fc2w, const float* __restrict__ fc2b,
                            float* __restrict__ out) {
    __shared__ float red[128];
    int g = blockIdx.x, t = threadIdx.x;
    float inv = 1.0f / fmaxf(cnt[g], 1.0f);
    float acc = fc1b[t];
    for (int k = 0; k < 128; ++k)
        acc += (pooled[g * 128 + k] * inv) * fc1w[k * 128 + t];
    red[t] = fmaxf(acc, 0.f) * fc2w[t];
    __syncthreads();
    for (int s = 64; s > 0; s >>= 1) {
        if (t < s) red[t] += red[t + s];
        __syncthreads();
    }
    if (t == 0) out[g] = red[0] + fc2b[0];
}

extern "C" void kernel_launch(void* const* d_in, const int* in_sizes, int n_in,
                              void* d_out, int out_size, void* d_ws, size_t ws_size,
                              hipStream_t stream) {
    const float* x         = (const float*)d_in[0];
    const int*   eix       = (const int*)d_in[1];   // int64 viewed as int pairs
    const float* edge_attr = (const float*)d_in[2];
    const int*   batch     = (const int*)d_in[3];
    const float* node_w = (const float*)d_in[4];  const float* node_b = (const float*)d_in[5];
    const float* ee_w   = (const float*)d_in[6];  const float* ee_b   = (const float*)d_in[7];
    const float* f1_w   = (const float*)d_in[8];  const float* f1_b   = (const float*)d_in[9];
    const float* f2_w   = (const float*)d_in[10]; const float* f2_b   = (const float*)d_in[11];
    const float* dn_w   = (const float*)d_in[12]; const float* dn_b   = (const float*)d_in[13];
    const float* me_w   = (const float*)d_in[14]; const float* me_b   = (const float*)d_in[15];
    const float* mn_w   = (const float*)d_in[16]; const float* mn_b   = (const float*)d_in[17];
    const float* fc1w   = (const float*)d_in[18]; const float* fc1b   = (const float*)d_in[19];
    const float* fc2w   = (const float*)d_in[20]; const float* fc2b   = (const float*)d_in[21];

    // workspace carve-up
    char* ws = (char*)d_ws;
    size_t off = 0;
    auto carve = [&](size_t bytes) { size_t o = off; off = (off + bytes + 255) & ~(size_t)255; return o; };
    _Float16* Pnode = (_Float16*)(ws + carve(8192 * 2));        // K=64
    _Float16* Pee   = (_Float16*)(ws + carve(4096 * 2));        // K=16->32
    _Float16* Pf1   = (_Float16*)(ws + carve(3 * 16384 * 2));   // K=128 x3
    _Float16* Pf2   = (_Float16*)(ws + carve(3 * 16384 * 2));
    _Float16* Pdn   = (_Float16*)(ws + carve(3 * 16384 * 2));
    _Float16* Pme   = (_Float16*)(ws + carve(3 * 16384 * 2));
    _Float16* Pmn   = (_Float16*)(ws + carve(3 * 32768 * 2));   // K=256 x3
    float* h      = (float*)(ws + carve((size_t)NN * 128 * 4));
    float* agg    = (float*)(ws + carve((size_t)NN * 128 * 4));
    float* agg2   = (float*)(ws + carve((size_t)NN * 256 * 4));
    float* pooled = (float*)(ws + carve((size_t)GG * 128 * 4));
    float* cnt    = (float*)(ws + carve((size_t)GG * 4));

    // pack weights into B-fragment layout (f16)
    auto packgrid = [](int total) { return (total + 255) / 256; };
    pack_b<<<packgrid(8192), 256, 0, stream>>>(node_w, 64, 2, Pnode);
    pack_b<<<packgrid(4096), 256, 0, stream>>>(ee_w, 16, 1, Pee);
    for (int k = 0; k < 3; ++k) {
        pack_b<<<packgrid(16384), 256, 0, stream>>>(f1_w + (size_t)k * 16384, 128, 4, Pf1 + (size_t)k * 16384);
        pack_b<<<packgrid(16384), 256, 0, stream>>>(f2_w + (size_t)k * 16384, 128, 4, Pf2 + (size_t)k * 16384);
        pack_b<<<packgrid(16384), 256, 0, stream>>>(dn_w + (size_t)k * 16384, 128, 4, Pdn + (size_t)k * 16384);
        pack_b<<<packgrid(16384), 256, 0, stream>>>(me_w + (size_t)k * 16384, 128, 4, Pme + (size_t)k * 16384);
        pack_b<<<packgrid(32768), 256, 0, stream>>>(mn_w + (size_t)k * 32768, 256, 8, Pmn + (size_t)k * 32768);
    }

    const int NTILES = NN / 16;                 // 1250
    const int nodeBlocks = (NTILES + 3) / 4;    // NW=4
    const int edgeBlocks = EE / 32 / 4;         // 32 edges/wave, 4 waves -> 5000

    // h = relu(x @ node_emb + b)
    node_update_kernel<64, 4><<<nodeBlocks, 128, 0, stream>>>(x, Pnode, node_b, h, NTILES);

    // SchNet layers
    for (int k = 0; k < 3; ++k) {
        hipMemsetAsync(agg, 0, (size_t)NN * 128 * 4, stream);
        schnet_edge_kernel<4><<<edgeBlocks, 128, 0, stream>>>(
            edge_attr, eix, h, agg, Pee, ee_b,
            Pf1 + (size_t)k * 16384, f1_b + k * 128,
            Pf2 + (size_t)k * 16384, f2_b + k * 128);
        node_update_kernel<128, 4><<<nodeBlocks, 128, 0, stream>>>(
            agg, Pdn + (size_t)k * 16384, dn_b + k * 128, h, NTILES);
    }

    // MPNN layers
    for (int k = 0; k < 3; ++k) {
        hipMemsetAsync(agg2, 0, (size_t)NN * 256 * 4, stream);
        mpnn_edge_kernel<4><<<edgeBlocks, 128, 0, stream>>>(
            edge_attr, eix, h, agg2, Pee, ee_b,
            Pme + (size_t)k * 16384, me_b + k * 128);
        node_update_kernel<256, 4><<<nodeBlocks, 128, 0, stream>>>(
            agg2, Pmn + (size_t)k * 32768, mn_b + k * 128, h, NTILES);
    }

    // mean pool + head
    hipMemsetAsync(pooled, 0, (size_t)GG * 128 * 4, stream);
    hipMemsetAsync(cnt, 0, (size_t)GG * 4, stream);
    pool_kernel<<<(NN * 128 + 255) / 256, 256, 0, stream>>>(h, batch, pooled, cnt);
    head_kernel<<<GG, 128, 0, stream>>>(pooled, cnt, fc1w, fc1b, fc2w, fc2b, (float*)d_out);
}
